// MultiAttention_78563541778951
// MI455X (gfx1250) — compile-verified, matrix operand order checked by reading, not run on hardware
//
#include <hip/hip_runtime.h>

// -------- CDNA5 (gfx1250) wave32 WMMA types --------
typedef __attribute__((ext_vector_type(16))) __bf16 v16bf;
typedef __attribute__((ext_vector_type(8)))  float  v8f;
typedef __attribute__((ext_vector_type(8)))  __bf16 bf16x8;
typedef __attribute__((ext_vector_type(8)))  float  f32x8;

// Problem sizes (fixed by the reference): S=1024, B=8, D=1024, H=8
#define SS 1024
#define BB 8
#define DD 1024
#define HH 8

constexpr int BM = 128, BN = 128, BK = 64;
constexpr int LDS_P = BK + 8;            // 72 halfs: 144B pitch = 9*16B (aligned b128, spread banks)
constexpr int TILE_H = BM * LDS_P;       // halfs per A/B tile buffer
constexpr size_t SMEM_BYTES = (size_t)4 * TILE_H * sizeof(__bf16);  // 2 bufs x (A+B) = 72 KB dynamic

// One wave-instruction: 32 lanes x 16B global -> LDS (ASYNCcnt-tracked, gfx1250 async path)
__device__ __forceinline__ void async_copy16(__bf16* l, const __bf16* g) {
  unsigned lds32 = (unsigned)(unsigned long long)l;       // low 32 bits of generic ptr = LDS offset
  unsigned long long ga = (unsigned long long)g;
  asm volatile("global_load_async_to_lds_b128 %0, %1, off"
               :: "v"(lds32), "v"(ga) : "memory");
}

// Stage one 128x64 bf16 tile pair into LDS buffers.
// A and K-major B: async global->LDS. Non-K-major B (V operand): VGPR transpose-scatter (DScnt).
template<bool B_KMAJ>
__device__ __forceinline__ void stage_tiles(int tid, __bf16* As, __bf16* Bs,
                                            const __bf16* __restrict__ A, long lda, long rowBase,
                                            const __bf16* __restrict__ B, long ldb, long colBase,
                                            int kt)
{
  #pragma unroll
  for (int i = 0; i < 4; ++i) {
    int ch = i * 256 + tid;
    int r  = ch >> 3;                  // 8 chunks of 8 halfs per row
    int c  = (ch & 7) * 8;
    async_copy16(As + r * LDS_P + c, A + (rowBase + r) * lda + kt + c);
  }
  if constexpr (B_KMAJ) {
    #pragma unroll
    for (int i = 0; i < 4; ++i) {
      int ch = i * 256 + tid;
      int r  = ch >> 3;
      int c  = (ch & 7) * 8;
      async_copy16(Bs + r * LDS_P + c, B + (colBase + r) * ldb + kt + c);
    }
  } else {
    #pragma unroll
    for (int i = 0; i < 4; ++i) {
      int ch = i * 256 + tid;
      int k  = ch >> 4;                // 16 chunks per k-row (BN/8)
      int n  = (ch & 15) * 8;
      bf16x8 v = *(const bf16x8*)(B + (long)(kt + k) * ldb + colBase + n);
      #pragma unroll
      for (int j = 0; j < 8; ++j) Bs[(n + j) * LDS_P + k] = v[j];
    }
  }
}

// C(M,N) = A(M,K) * B^T + bias; bf16 WMMA, f32 accum; double-buffered async LDS pipeline.
// grid: (N/BN, M/BM, batch); batch advances A/B/C by {a,b,c}Batch elements.
template<bool B_KMAJ, bool BIAS, bool ACCUM, bool OUT_BF16>
__global__ __launch_bounds__(256)
void gemm_wmma(const __bf16* __restrict__ A, long lda, long aBatch,
               const __bf16* __restrict__ B, long ldb, long bBatch,
               const float* __restrict__ bias,
               void* __restrict__ Cv, long ldc, long cBatch,
               int K)
{
  extern __shared__ __align__(16) __bf16 smem[];
  // layout: [Abuf0][Abuf1][Bbuf0][Bbuf1]
  constexpr int ASYNC_PER_STAGE = 4 + (B_KMAJ ? 4 : 0);   // per-wave async instrs per stage

  const int tid  = threadIdx.x;
  const int lane = tid & 31;
  const int wave = tid >> 5;     // 8 waves
  const int wm   = wave >> 2;    // 0..1 -> 64-row slab
  const int wn   = wave & 3;     // 0..3 -> 32-col slab
  const int lr   = lane & 15;    // M/N index inside 16x16 tile
  const int hi   = lane >> 4;    // K-subgroup select (ISA 16-bit A/B layout)

  const long rowBase = (long)blockIdx.y * BM;
  const long colBase = (long)blockIdx.x * BN;
  A += (long)blockIdx.z * aBatch;
  B += (long)blockIdx.z * bBatch;

  v8f acc[4][2] = {};

  stage_tiles<B_KMAJ>(tid, smem, smem + 2 * TILE_H, A, lda, rowBase, B, ldb, colBase, 0);

  int buf = 0;
  for (int kt = 0; kt < K; kt += BK) {
    const bool hasNext = (kt + BK) < K;
    if (hasNext)
      stage_tiles<B_KMAJ>(tid, smem + (buf ^ 1) * TILE_H, smem + (2 + (buf ^ 1)) * TILE_H,
                          A, lda, rowBase, B, ldb, colBase, kt + BK);
    // In-order async completion: <= ASYNC_PER_STAGE outstanding => current buffer is in LDS.
    if (hasNext) asm volatile("s_wait_asynccnt %0" :: "i"(ASYNC_PER_STAGE) : "memory");
    else         asm volatile("s_wait_asynccnt 0" ::: "memory");
    __syncthreads();                       // all waves' staging of `buf` visible (+DScnt for scatter)

    const __bf16* Ab = smem + buf * TILE_H;
    const __bf16* Bb = smem + (2 + buf) * TILE_H;
    #pragma unroll
    for (int ks = 0; ks < BK; ks += 32) {
      v16bf fa[4], fb[2];
      #pragma unroll
      for (int mt = 0; mt < 4; ++mt) {
        const __bf16* p = &Ab[(wm * 64 + mt * 16 + lr) * LDS_P + ks + hi * 8];
        bf16x8 lo = *(const bf16x8*)p;            // K = ks+8*hi   .. +7
        bf16x8 hq = *(const bf16x8*)(p + 16);     // K = ks+16+8*hi.. +7
        fa[mt] = __builtin_shufflevector(lo, hq, 0,1,2,3,4,5,6,7,8,9,10,11,12,13,14,15);
      }
      #pragma unroll
      for (int nt = 0; nt < 2; ++nt) {
        const __bf16* p = &Bb[(wn * 32 + nt * 16 + lr) * LDS_P + ks + hi * 8];
        bf16x8 lo = *(const bf16x8*)p;
        bf16x8 hq = *(const bf16x8*)(p + 16);
        fb[nt] = __builtin_shufflevector(lo, hq, 0,1,2,3,4,5,6,7,8,9,10,11,12,13,14,15);
      }
      #pragma unroll
      for (int mt = 0; mt < 4; ++mt)
        #pragma unroll
        for (int nt = 0; nt < 2; ++nt)
          acc[mt][nt] = __builtin_amdgcn_wmma_f32_16x16x32_bf16(
              false, fa[mt], false, fb[nt], (short)0, acc[mt][nt], false, false);
    }
    __syncthreads();                       // WAR: done reading `buf` before it is restaged
    buf ^= 1;
  }

  // epilogue: v8f element r -> (M = r + 8*hi, N = lr)
  #pragma unroll
  for (int mt = 0; mt < 4; ++mt) {
    #pragma unroll
    for (int nt = 0; nt < 2; ++nt) {
      const long col = colBase + wn * 32 + nt * 16 + lr;
      const float bv = BIAS ? bias[col] : 0.0f;
      #pragma unroll
      for (int r = 0; r < 8; ++r) {
        const long row = rowBase + wm * 64 + mt * 16 + hi * 8 + r;
        float v = acc[mt][nt][r] + bv;
        if constexpr (OUT_BF16) {
          __bf16* C = (__bf16*)Cv + (long)blockIdx.z * cBatch;
          C[row * ldc + col] = (__bf16)v;
        } else {
          float* C = (float*)Cv + (long)blockIdx.z * cBatch;
          if constexpr (ACCUM) v += C[row * ldc + col];
          C[row * ldc + col] = v;
        }
      }
    }
  }
}

// One-shot f32 -> bf16 conversion (runs once per call; ~160MB reads => ~7us at 23.3 TB/s)
__global__ __launch_bounds__(256)
void f32_to_bf16(const float* __restrict__ src, __bf16* __restrict__ dst, long n)
{
  long i = ((long)blockIdx.x * 256 + threadIdx.x) * 8;
  if (i < n) {
    f32x8 f = *(const f32x8*)(src + i);
    *(bf16x8*)(dst + i) = __builtin_convertvector(f, bf16x8);
  }
}

// Row softmax over 1024 f32 -> bf16; scale folded: softmax(scale*s).
__global__ __launch_bounds__(256)
void softmax_rows(const float* __restrict__ Sc, __bf16* __restrict__ At, float scale)
{
  const long row = blockIdx.x;
  const float* rp = Sc + row * (long)SS;
  __bf16* ap = At + row * (long)SS;
  const int t = threadIdx.x;

  float4 v = ((const float4*)rp)[t];
  float m = fmaxf(fmaxf(v.x, v.y), fmaxf(v.z, v.w));
  #pragma unroll
  for (int off = 16; off > 0; off >>= 1) m = fmaxf(m, __shfl_xor(m, off, 32));

  __shared__ float redm[8], reds[8];
  const int wv = t >> 5, ln = t & 31;
  if (ln == 0) redm[wv] = m;
  __syncthreads();
  m = redm[0];
  #pragma unroll
  for (int i = 1; i < 8; ++i) m = fmaxf(m, redm[i]);

  float e0 = __expf((v.x - m) * scale);
  float e1 = __expf((v.y - m) * scale);
  float e2 = __expf((v.z - m) * scale);
  float e3 = __expf((v.w - m) * scale);
  float s = e0 + e1 + e2 + e3;
  #pragma unroll
  for (int off = 16; off > 0; off >>= 1) s += __shfl_xor(s, off, 32);
  if (ln == 0) reds[wv] = s;
  __syncthreads();
  s = reds[0];
  #pragma unroll
  for (int i = 1; i < 8; ++i) s += reds[i];
  const float inv = 1.0f / s;

  struct __align__(8) BF4 { __bf16 v[4]; } o;
  o.v[0] = (__bf16)(e0 * inv); o.v[1] = (__bf16)(e1 * inv);
  o.v[2] = (__bf16)(e2 * inv); o.v[3] = (__bf16)(e3 * inv);
  *(BF4*)(ap + t * 4) = o;
}

extern "C" void kernel_launch(void* const* d_in, const int* in_sizes, int n_in,
                              void* d_out, int out_size, void* d_ws, size_t ws_size,
                              hipStream_t stream) {
  const float* x     = (const float*)d_in[0];   // (S,B,D)
  const float* w_in  = (const float*)d_in[1];   // (H,3D,D)
  const float* b_in  = (const float*)d_in[2];   // (H,3D)
  const float* w_out = (const float*)d_in[3];   // (H,D,D)
  const float* b_out = (const float*)d_in[4];   // (H,D)
  float* out = (float*)d_out;                   // (S,B,D) f32

  // workspace partition (192 MB total)
  char* w = (char*)d_ws;
  __bf16* xb  = (__bf16*)w;  w += (size_t)SS * BB * DD * sizeof(__bf16);          // 16 MB
  __bf16* wib = (__bf16*)w;  w += (size_t)HH * 3 * DD * DD * sizeof(__bf16);      // 48 MB
  __bf16* wob = (__bf16*)w;  w += (size_t)HH * DD * DD * sizeof(__bf16);          // 16 MB
  __bf16* qkv = (__bf16*)w;  w += (size_t)SS * BB * 3 * DD * sizeof(__bf16);      // 48 MB
  float*  sc  = (float*)w;   w += (size_t)BB * SS * SS * sizeof(float);           // 32 MB
  __bf16* at  = (__bf16*)w;  w += (size_t)BB * SS * SS * sizeof(__bf16);          // 16 MB
  __bf16* ov  = (__bf16*)w;  w += (size_t)BB * SS * DD * sizeof(__bf16);          // 16 MB

  const long M  = (long)SS * BB;      // 8192 flattened (s,b) rows
  const long N3 = 3 * DD;             // 3072
  const dim3 blk(256);
  const float scale = 0.03125f;       // 1/sqrt(1024)

  // Pre-convert all f32 operands to bf16 so every GEMM stage uses async-to-LDS.
  { long n = (long)SS * BB * DD;
    f32_to_bf16<<<dim3((unsigned)(n / (8 * 256))), blk, 0, stream>>>(x, xb, n); }
  { long n = (long)HH * N3 * DD;
    f32_to_bf16<<<dim3((unsigned)(n / (8 * 256))), blk, 0, stream>>>(w_in, wib, n); }
  { long n = (long)HH * DD * DD;
    f32_to_bf16<<<dim3((unsigned)(n / (8 * 256))), blk, 0, stream>>>(w_out, wob, n); }

  for (int h = 0; h < HH; ++h) {
    // 1) qkv = x * w_in[h]^T + b_in[h]  -> bf16
    gemm_wmma<true, true, false, true>
      <<<dim3(N3 / BN, M / BM, 1), blk, SMEM_BYTES, stream>>>(
        xb, DD, 0,
        wib + (size_t)h * N3 * DD, DD, 0,
        b_in + (size_t)h * N3,
        qkv, N3, 0, DD);

    // 2) scores[b] = q_b * k_b^T  (strided rows of qkv), f32 out
    gemm_wmma<true, false, false, false>
      <<<dim3(SS / BN, SS / BM, BB), blk, SMEM_BYTES, stream>>>(
        qkv,      (long)BB * N3, N3,
        qkv + DD, (long)BB * N3, N3,
        nullptr,
        sc, SS, (long)SS * SS, DD);

    // 3) softmax rows (scale folded), bf16 attn
    softmax_rows<<<BB * SS, blk, 0, stream>>>(sc, at, scale);

    // 4) o_b = attn_b * v_b   (v is (K=t, N=e): transpose-staged into LDS)
    gemm_wmma<false, false, false, true>
      <<<dim3(DD / BN, SS / BM, BB), blk, SMEM_BYTES, stream>>>(
        at,           SS, (long)SS * SS,
        qkv + 2 * DD, (long)BB * N3, N3,
        nullptr,
        ov, DD, (long)SS * DD, SS);

    // 5) out[b] += o_b * w_out[h]^T + b_out[h]  (f32 accumulate across heads)
    if (h == 0)
      gemm_wmma<true, true, false, false>
        <<<dim3(DD / BN, SS / BM, BB), blk, SMEM_BYTES, stream>>>(
          ov, DD, (long)SS * DD,
          wob + (size_t)h * DD * DD, DD, 0,
          b_out + (size_t)h * DD,
          out, (long)BB * DD, DD, DD);
    else
      gemm_wmma<true, true, true, false>
        <<<dim3(DD / BN, SS / BM, BB), blk, SMEM_BYTES, stream>>>(
          ov, DD, (long)SS * DD,
          wob + (size_t)h * DD * DD, DD, 0,
          b_out + (size_t)h * DD,
          out, (long)BB * DD, DD, DD);
  }
}
